// TRUNET_CGRU_Input_Layer_90606630077081
// MI455X (gfx1250) — matrix operand order repaired and checked by reading, not compiled, not run on hardware
//
#include <hip/hip_runtime.h>

// ---------------- CDNA5 WMMA / TDM types ----------------
typedef __attribute__((ext_vector_type(16))) _Float16 v16h;
typedef __attribute__((ext_vector_type(8)))  float    v8f;
typedef unsigned int u32x4 __attribute__((ext_vector_type(4)));
typedef int          i32x8 __attribute__((ext_vector_type(8)));
typedef int          i32x4 __attribute__((ext_vector_type(4)));

#define DEVINL __device__ __forceinline__

namespace {

constexpr int Bn = 2, Tn = 12, Hn = 80, Wn = 120, Cin = 6;
constexpr int PIX  = Hn * Wn;        // 9600 pixels per image
constexpr int PTOT = Bn * Tn * PIX;  // 230400 = input-conv GEMM M (multiple of 64)
constexpr int PBH  = Bn * PIX;       // 19200  = recurrent GEMM M (multiple of 64)

// input conv GEMM: K = 3*3*6 = 54 -> padded to 64; N = 2 dirs * 96 = 192
constexpr int KX   = 64;
constexpr int NX   = 192;
constexpr int LDBX = 72;   // padded LDS row (halves): 144B, 16B aligned, bank-skewed

// recurrent GEMMs: K = 3*3*32 = 288, staged in K-panels of 96 (one filter row)
constexpr int KREC   = 288;
constexpr int KPANEL = 96;
constexpr int LDP    = 104; // padded A-panel LDS row (halves): 208B, 16B aligned
// TDM-filled weight panels are stored dense: row stride = KPANEL (192B, 16B aligned)

union H8 { uint4 u; _Float16 h[8]; };

// A fragment 16x32 (MxK), 16-bit (ISA 7.12.2):
//   lanes 0-15 : M=l, K in {kk+0..7,  kk+16..23}
//   lanes 16-31: M=l, K in {kk+8..15, kk+24..31}
DEVINL v16h frag_a(const _Float16* sA, int lda, int mtile, int kk, int lane) {
  const int l = lane & 15, hi = lane >> 4;
  const _Float16* row = sA + (mtile * 16 + l) * lda + kk + hi * 8;
  H8 c0, c1;
  c0.u = *(const uint4*)(row);
  c1.u = *(const uint4*)(row + 16);
  v16h a;
#pragma unroll
  for (int i = 0; i < 8; ++i) { a[i] = c0.h[i]; a[8 + i] = c1.h[i]; }
  return a;
}

// B fragment 32x16 (KxN), 16-bit, weights staged N-major ([n][k]):
//   lanes 0-15 : N=l, K in {kk+0..7, kk+16..23}; lanes 16-31: N=l, K in {kk+8..15, kk+24..31}
DEVINL v16h frag_b(const _Float16* sW, int ldb, int ntile, int kk, int lane) {
  const int l = lane & 15, hi = lane >> 4;
  const _Float16* row = sW + (ntile * 16 + l) * ldb + kk + hi * 8;
  H8 c0, c1;
  c0.u = *(const uint4*)(row);
  c1.u = *(const uint4*)(row + 16);
  v16h b;
#pragma unroll
  for (int i = 0; i < 8; ++i) { b[i] = c0.h[i]; b[8 + i] = c1.h[i]; }
  return b;
}

DEVINL float sigmoidf_(float v) { return 1.0f / (1.0f + __expf(-v)); }

// Generic pointer to a __shared__ object: low 32 bits are the LDS byte address.
DEVINL unsigned int lds_addr_of(const void* p) {
  return (unsigned int)(unsigned long long)p;
}

// Async global->LDS 16B copy (ASYNCcnt path). Inline asm: portable across the
// two toolchains (clang builtin arity differs). saddr form: vdst = LDS byte
// address, vaddr = 32-bit byte offset, saddr = 64-bit base.
DEVINL void async_ld_b128(unsigned int lds_byte, unsigned int goff_byte,
                          const void* gbase) {
  asm volatile("global_load_async_to_lds_b128 %0, %1, %2"
               :
               : "v"(lds_byte), "v"(goff_byte), "s"(gbase)
               : "memory");
}
DEVINL void wait_asynccnt0() {
  asm volatile("s_wait_asynccnt 0x0" ::: "memory");
}

// Tensor Data Mover: DMA a 2D f16 tile [tile_d1 x tile_d0] out of a tensor
// [tensor_d1 x tensor_d0] (row stride `stride`, elements) into LDS at `lds`.
// D# layout per CDNA5 ISA 8.3/8.4 (group0: count/lds/global/type; group1:
// data_size=2B, dims, tile dims, dim0 stride). Groups 2/3 zero (2D tensor).
DEVINL void tdm_load_2d_f16(unsigned int lds, const _Float16* gptr,
                            int tensor_d0, int tensor_d1,
                            int tile_d0, int tile_d1, int stride) {
  const unsigned long long ga = (unsigned long long)gptr;
  u32x4 g0;
  g0[0] = 1u;                                   // count=1 valid descriptor
  g0[1] = lds;                                  // lds_addr
  g0[2] = (unsigned int)ga;                     // global_addr[31:0]
  g0[3] = (unsigned int)(ga >> 32) | (2u << 30);// global_addr[56:32] | type=2
  i32x8 g1;
  g1[0] = (1 << 16);                            // data_size = 1 -> 2 bytes
  g1[1] = (tensor_d0 & 0xFFFF) << 16;           // tensor_dim0[15:0]
  g1[2] = (tensor_d0 >> 16) | ((tensor_d1 & 0xFFFF) << 16);
  g1[3] = (tensor_d1 >> 16) | (tile_d0 << 16);  // tile_dim0
  g1[4] = tile_d1;                              // tile_dim1 (tile_dim2=0)
  g1[5] = stride;                               // tensor_dim0_stride[31:0]
  g1[6] = 0;
  g1[7] = 0;
  i32x4 z4 = {0, 0, 0, 0};
#if defined(__clang_major__) && __clang_major__ >= 23
  i32x8 z8 = {};
  __builtin_amdgcn_tensor_load_to_lds(g0, g1, z4, z4, z8, 0);
#else
  __builtin_amdgcn_tensor_load_to_lds(g0, g1, z4, z4, 0);
#endif
}

} // namespace

// -------- weight repack: f32 HWIO -> f16 N-major [n][k] --------
__global__ void wconv_kernel(
    const float* __restrict__ Wxf, const float* __restrict__ Wxb,
    const float* __restrict__ Wzrf, const float* __restrict__ Wzrb,
    const float* __restrict__ Whhf, const float* __restrict__ Whhb,
    _Float16* __restrict__ Wxp, _Float16* __restrict__ Wzrp,
    _Float16* __restrict__ Whhp) {
  const int tid = blockIdx.x * blockDim.x + threadIdx.x;
  const int stride = gridDim.x * blockDim.x;
  for (int idx = tid; idx < NX * KX; idx += stride) {
    const int n = idx >> 6, k = idx & 63;
    const int dir = n >= 96 ? 1 : 0, co = n - dir * 96;
    float v = 0.0f;
    if (k < 54) v = (dir ? Wxb : Wxf)[k * 96 + co];
    Wxp[idx] = (_Float16)v;
  }
  for (int idx = tid; idx < 2 * 64 * KREC; idx += stride) {
    const int dir = idx / (64 * KREC), r = idx % (64 * KREC);
    const int n = r / KREC, k = r % KREC;
    Wzrp[idx] = (_Float16)((dir ? Wzrb : Wzrf)[k * 64 + n]);
  }
  for (int idx = tid; idx < 2 * 32 * KREC; idx += stride) {
    const int dir = idx / (32 * KREC), r = idx % (32 * KREC);
    const int n = r / KREC, k = r % KREC;
    Whhp[idx] = (_Float16)((dir ? Whhb : Whhf)[k * 32 + n]);
  }
}

__global__ void init_h_kernel(float* __restrict__ h, int n) {
  const int i = blockIdx.x * blockDim.x + threadIdx.x;
  if (i < n) h[i] = 0.0f;
}

// -------- input conv: implicit GEMM, M=230400, K=54->64, N=192 (both dirs) --------
__global__ __launch_bounds__(256) void convx_kernel(
    const float* __restrict__ x, const _Float16* __restrict__ Wxp,
    const float* __restrict__ bf, const float* __restrict__ bb,
    _Float16* __restrict__ xc) {
  __shared__ alignas(16) _Float16 sA[64 * KX];
  __shared__ alignas(16) _Float16 sW[NX * LDBX];
  __shared__ float sBias[NX];

  const int tid  = threadIdx.x;
  const int base = blockIdx.x * 64;

  // async-stage packed weights [n][k] into padded LDS rows: 1536 x 16B chunks
  {
    const unsigned int wbase = lds_addr_of(sW);
#pragma unroll
    for (int it = 0; it < 6; ++it) {
      const int c = tid + it * 256;          // 1536 = 6*256 exactly
      const int n = c >> 3, k8 = (c & 7) * 8;
      async_ld_b128(wbase + (unsigned)(n * LDBX + k8) * 2,
                    (unsigned)(n * 64 + k8) * 2, Wxp);
    }
  }

  // stage input patches (f32 -> f16), SAME-padding halo, K padded 54..63 = 0
#pragma unroll 4
  for (int idx = tid; idx < 64 * KX; idx += 256) {
    const int m = idx >> 6, k = idx & 63;
    float v = 0.0f;
    if (k < 54) {
      const int q = base + m;
      const int img = q / PIX, rem = q % PIX;
      const int hy = rem / Wn, wx = rem % Wn;
      const int ky = k / 18, kx = (k / 6) % 3, ci = k % 6;
      const int hs = hy + ky - 1, wsx = wx + kx - 1;
      if ((unsigned)hs < (unsigned)Hn && (unsigned)wsx < (unsigned)Wn)
        v = x[(((size_t)img * Hn + hs) * Wn + wsx) * Cin + ci];
    }
    sA[m * KX + k] = (_Float16)v;
  }
  if (tid < NX) sBias[tid] = (tid < 96) ? bf[tid] : bb[tid - 96];
  wait_asynccnt0();
  __syncthreads();

  const int wave = tid >> 5, lane = tid & 31;  // wave32
  const int mtile = wave & 3;         // 4 m-tiles of 16 pixels
  const int nt0   = (wave >> 2) * 6;  // 12 n-tiles over 2 wave-columns

  v8f zero = {};
  v8f acc[6];
#pragma unroll
  for (int j = 0; j < 6; ++j) acc[j] = zero;

#pragma unroll
  for (int kk = 0; kk < KX; kk += 32) {
    const v16h a = frag_a(sA, KX, mtile, kk, lane);
#pragma unroll
    for (int j = 0; j < 6; ++j) {
      const v16h b = frag_b(sW, LDBX, nt0 + j, kk, lane);
      acc[j] = __builtin_amdgcn_wmma_f32_16x16x32_f16(
          false, a, false, b, (short)0, acc[j], false, false);
    }
  }

  // epilogue: +bias, store f16 xc[dir][pixel][96]
  const int l = lane & 15, hi = lane >> 4;
#pragma unroll
  for (int j = 0; j < 6; ++j) {
    const int n   = (nt0 + j) * 16 + l;
    const int dir = n >= 96 ? 1 : 0;
    const int ch  = n - dir * 96;
    _Float16* dst = xc + (size_t)dir * PTOT * 96 + ch;
    const float bias = sBias[n];
#pragma unroll
    for (int v = 0; v < 8; ++v) {
      const int q = base + mtile * 16 + hi * 8 + v;
      dst[(size_t)q * 96] = (_Float16)(acc[j][v] + bias);
    }
  }
}

// -------- GRU gates: hzr = conv(h, Wzr); z = sig(xz+hz); rh = sig(xr+hr)*h --------
__global__ __launch_bounds__(256) void gate_kernel(
    const float* __restrict__ hstate, const _Float16* __restrict__ xc_dir,
    const _Float16* __restrict__ Wzr, _Float16* __restrict__ zbuf,
    _Float16* __restrict__ rhbuf, int t) {
  __shared__ alignas(16) _Float16 sA[64 * LDP];
  __shared__ alignas(16) _Float16 sW[64 * KPANEL];  // TDM-filled, dense rows

  const int tid  = threadIdx.x;
  const int base = blockIdx.x * 64;
  const int wave = tid >> 5, lane = tid & 31;
  const int mtile = wave & 3;
  const int nt0   = (wave >> 2) * 2;

  {  // speculative prefetch of this block's x-conv slab (global_prefetch_b8)
    const int q = base + (tid & 63);
    const int b_ = q / PIX, hw = q % PIX;
    __builtin_prefetch(xc_dir + ((size_t)(b_ * Tn + t) * PIX + hw) * 96, 0, 0);
  }

  v8f zero = {};
  v8f acc[2]; acc[0] = zero; acc[1] = zero;

  for (int kp = 0; kp < KREC; kp += KPANEL) {
    __syncthreads();
    // TDM: DMA weight panel [64 x 96] f16 (tensor [64 x 288]) into LDS
    if (wave == 0)
      tdm_load_2d_f16(lds_addr_of(sW), Wzr + kp, KREC, 64, KPANEL, 64, KREC);
    const int ky = kp / KPANEL;  // filter row of this K-panel
#pragma unroll 2
    for (int idx = tid; idx < 64 * KPANEL; idx += 256) {
      const int m = idx / KPANEL, kloc = idx % KPANEL;
      const int kx = kloc >> 5, ci = kloc & 31;
      const int q = base + m;
      const int b_ = q / PIX, rem = q % PIX;
      const int hy = rem / Wn, wx = rem % Wn;
      const int hs = hy + ky - 1, wsx = wx + kx - 1;
      float v = 0.0f;
      if ((unsigned)hs < (unsigned)Hn && (unsigned)wsx < (unsigned)Wn)
        v = hstate[((size_t)b_ * PIX + hs * Wn + wsx) * 32 + ci];
      sA[m * LDP + kloc] = (_Float16)v;
    }
    if (wave == 0) __builtin_amdgcn_s_wait_tensorcnt(0);
    __syncthreads();
#pragma unroll
    for (int kk = 0; kk < KPANEL; kk += 32) {
      const v16h a = frag_a(sA, LDP, mtile, kk, lane);
#pragma unroll
      for (int j = 0; j < 2; ++j) {
        const v16h b = frag_b(sW, KPANEL, nt0 + j, kk, lane);
        acc[j] = __builtin_amdgcn_wmma_f32_16x16x32_f16(
            false, a, false, b, (short)0, acc[j], false, false);
      }
    }
  }

  const int l = lane & 15, hi = lane >> 4;
#pragma unroll
  for (int j = 0; j < 2; ++j) {
    const int n = (nt0 + j) * 16 + l;  // 0..63
#pragma unroll
    for (int v = 0; v < 8; ++v) {
      const int q  = base + mtile * 16 + hi * 8 + v;
      const int b_ = q / PIX, hw = q % PIX;
      const size_t xcpix = (size_t)(b_ * Tn + t) * PIX + hw;
      const float g = sigmoidf_((float)xc_dir[xcpix * 96 + n] + acc[j][v]);
      if (n < 32) {
        zbuf[(size_t)q * 32 + n] = (_Float16)g;
      } else {
        const int c = n - 32;
        rhbuf[(size_t)q * 32 + c] = (_Float16)(g * hstate[(size_t)q * 32 + c]);
      }
    }
  }
}

// -------- candidate + update: hh = tanh(xh + conv(rh, Whh)); h = z*h+(1-z)*hh --------
__global__ __launch_bounds__(256) void cand_kernel(
    float* __restrict__ hstate, const _Float16* __restrict__ xc_dir,
    const _Float16* __restrict__ Whh, const _Float16* __restrict__ zbuf,
    const _Float16* __restrict__ rhbuf, float* __restrict__ out,
    int t, int dir) {
  __shared__ alignas(16) _Float16 sA[64 * LDP];
  __shared__ alignas(16) _Float16 sW[32 * KPANEL];  // TDM-filled, dense rows

  const int tid  = threadIdx.x;
  const int base = blockIdx.x * 64;
  const int wave = tid >> 5, lane = tid & 31;
  const int mtile = wave & 3;
  const int ntile = wave >> 2;  // 0..1

  {
    const int q = base + (tid & 63);
    __builtin_prefetch(zbuf + (size_t)q * 32, 0, 0);
  }

  v8f acc = {};
  for (int kp = 0; kp < KREC; kp += KPANEL) {
    __syncthreads();
    if (wave == 0)
      tdm_load_2d_f16(lds_addr_of(sW), Whh + kp, KREC, 32, KPANEL, 32, KREC);
    const int ky = kp / KPANEL;
#pragma unroll 2
    for (int idx = tid; idx < 64 * KPANEL; idx += 256) {
      const int m = idx / KPANEL, kloc = idx % KPANEL;
      const int kx = kloc >> 5, ci = kloc & 31;
      const int q = base + m;
      const int b_ = q / PIX, rem = q % PIX;
      const int hy = rem / Wn, wx = rem % Wn;
      const int hs = hy + ky - 1, wsx = wx + kx - 1;
      _Float16 v = (_Float16)0.0f;
      if ((unsigned)hs < (unsigned)Hn && (unsigned)wsx < (unsigned)Wn)
        v = rhbuf[((size_t)b_ * PIX + hs * Wn + wsx) * 32 + ci];
      sA[m * LDP + kloc] = v;
    }
    if (wave == 0) __builtin_amdgcn_s_wait_tensorcnt(0);
    __syncthreads();
#pragma unroll
    for (int kk = 0; kk < KPANEL; kk += 32) {
      const v16h a = frag_a(sA, LDP, mtile, kk, lane);
      const v16h b = frag_b(sW, KPANEL, ntile, kk, lane);
      acc = __builtin_amdgcn_wmma_f32_16x16x32_f16(
          false, a, false, b, (short)0, acc, false, false);
    }
  }

  const int l = lane & 15, hi = lane >> 4;
  const int n = ntile * 16 + l;  // 0..31
#pragma unroll
  for (int v = 0; v < 8; ++v) {
    const int q  = base + mtile * 16 + hi * 8 + v;
    const int b_ = q / PIX, hw = q % PIX;
    const size_t xcpix = (size_t)(b_ * Tn + t) * PIX + hw;
    const float xh = (float)xc_dir[xcpix * 96 + 64 + n];
    const float hh = tanhf(xh + acc[v]);
    const float z  = (float)zbuf[(size_t)q * 32 + n];
    const float ho = hstate[(size_t)q * 32 + n];
    const float hn = z * ho + (1.0f - z) * hh;
    hstate[(size_t)q * 32 + n] = hn;
    out[xcpix * 64 + (size_t)dir * 32 + n] = hn;
  }
}

extern "C" void kernel_launch(void* const* d_in, const int* in_sizes, int n_in,
                              void* d_out, int out_size, void* d_ws, size_t ws_size,
                              hipStream_t stream) {
  const float* x     = (const float*)d_in[0];
  const float* Wx_f  = (const float*)d_in[1];
  const float* b_f   = (const float*)d_in[2];
  const float* Wzr_f = (const float*)d_in[3];
  const float* Whh_f = (const float*)d_in[4];
  const float* Wx_b  = (const float*)d_in[5];
  const float* b_b   = (const float*)d_in[6];
  const float* Wzr_b = (const float*)d_in[7];
  const float* Whh_b = (const float*)d_in[8];
  float* out = (float*)d_out;
  (void)in_sizes; (void)n_in; (void)out_size; (void)ws_size;

  char* ws = (char*)d_ws;
  size_t off = 0;
  auto take = [&](size_t bytes) -> char* {
    char* p = ws + off;
    off = (off + bytes + 255) & ~(size_t)255;
    return p;
  };

  _Float16* xc   = (_Float16*)take((size_t)2 * PTOT * 96 * sizeof(_Float16)); // 88.5 MB
  float*    hst  = (float*)   take((size_t)2 * PBH * 32 * sizeof(float));     //  4.9 MB
  _Float16* zbuf = (_Float16*)take((size_t)2 * PBH * 32 * sizeof(_Float16));
  _Float16* rhb  = (_Float16*)take((size_t)2 * PBH * 32 * sizeof(_Float16));
  _Float16* Wxp  = (_Float16*)take((size_t)NX * KX * sizeof(_Float16));
  _Float16* Wzrp = (_Float16*)take((size_t)2 * 64 * KREC * sizeof(_Float16));
  _Float16* Whhp = (_Float16*)take((size_t)2 * 32 * KREC * sizeof(_Float16));

  wconv_kernel<<<64, 256, 0, stream>>>(Wx_f, Wx_b, Wzr_f, Wzr_b, Whh_f, Whh_b,
                                       Wxp, Wzrp, Whhp);
  const int hcount = 2 * PBH * 32;
  init_h_kernel<<<(hcount + 255) / 256, 256, 0, stream>>>(hst, hcount);
  convx_kernel<<<PTOT / 64, 256, 0, stream>>>(x, Wxp, b_f, b_b, xc);

  const size_t xcDir = (size_t)PTOT * 96;
  float*    hF = hst;  float*    hB = hst  + (size_t)PBH * 32;
  _Float16* zF = zbuf; _Float16* zB = zbuf + (size_t)PBH * 32;
  _Float16* rF = rhb;  _Float16* rB = rhb  + (size_t)PBH * 32;

  const int gridP = PBH / 64;  // 300 blocks
  for (int s = 0; s < Tn; ++s) {
    const int tf = s, tb = Tn - 1 - s;
    gate_kernel<<<gridP, 256, 0, stream>>>(hF, xc, Wzrp, zF, rF, tf);
    cand_kernel<<<gridP, 256, 0, stream>>>(hF, xc, Whhp, zF, rF, out, tf, 0);
    gate_kernel<<<gridP, 256, 0, stream>>>(hB, xc + xcDir, Wzrp + (size_t)64 * KREC,
                                           zB, rB, tb);
    cand_kernel<<<gridP, 256, 0, stream>>>(hB, xc + xcDir, Whhp + (size_t)32 * KREC,
                                           zB, rB, out, tb, 1);
  }
}
